// MultiAttention_18494129176678
// MI455X (gfx1250) — matrix operand order repaired and checked by reading, not compiled, not run on hardware
//
#include <hip/hip_runtime.h>
#include <hip/hip_bf16.h>

#define BB 4
#define SS 1024
#define DD 64
#define BQ 64      // queries per workgroup
#define KT 32      // keys per inner iteration
#define NW 4       // waves per workgroup

typedef __attribute__((ext_vector_type(16))) __bf16 v16bf;
typedef __attribute__((ext_vector_type(8)))  float  v8f;

// Native CDNA5 v_tanh_f32 if the toolchain exposes it; otherwise
// tanh(x) = 1 - 2/(e^{2x}+1) via v_exp_f32 + v_rcp_f32.
__device__ __forceinline__ float dev_tanh(float x) {
#if defined(__gfx1250__) && __has_builtin(__builtin_amdgcn_tanhf)
    return __builtin_amdgcn_tanhf(x);
#else
    float e = __expf(2.0f * x);
    return 1.0f - 2.0f * __builtin_amdgcn_rcpf(e + 1.0f);
#endif
}

// DS ops are in-order within a wave (ISA §7.3); this stops the compiler from
// hoisting the A-fragment loads above the P stores.
__device__ __forceinline__ void lds_fence_wave() {
    asm volatile("s_wait_dscnt 0" ::: "memory");
}

// 16-bit A fragment, 16x32 (M x K), contiguous K order (matches B frags):
// lanes 0-15: row M=lane, VGPR0-7 K=0..15
// lanes16-31: row M=lane-16, VGPR0-7 K=16..31
__device__ __forceinline__ v16bf wmma_a_frag(const __bf16* base, int ldr, int lane) {
    int m   = lane & 15;
    int kb0 = (lane < 16) ? 0 : 16;
    const __bf16* r = base + m * ldr;
    v16bf a;
#pragma unroll
    for (int v = 0; v < 8; ++v) {
        int k = kb0 + 2 * v;
        a[2 * v]     = r[k];
        a[2 * v + 1] = r[k + 1];
    }
    return a;
}

// 16-bit B fragment, 32x16 (K x N): lane holds column n=lane&15,
// lanes 0-15 cover K=0..15, lanes 16-31 cover K=16..31 (pairs per VGPR).
// With K pairs contiguous in the LDS tile this is one 32B ds read.
__device__ __forceinline__ v16bf wmma_b_frag_rm(const __bf16* tile, int ld,
                                                int n0, int koff, int lane) {
    int nn = n0 + (lane & 15);
    int kb = (lane < 16) ? 0 : 16;
    return *(const v16bf*)(tile + (size_t)nn * ld + koff + kb);
}

__device__ __forceinline__ v8f wmma_bf16(v16bf a, v16bf b, v8f c) {
    return __builtin_amdgcn_wmma_f32_16x16x32_bf16(
        /*neg_a=*/false, a, /*neg_b=*/false, b,
        /*c_mod=*/(short)0, c, /*reuse_a=*/false, /*reuse_b=*/false);
}

// Online-softmax update: row stats reduced across the 16-lane half that holds
// each row of the C tiles (4 x shfl_xor), accumulators rescaled.
__device__ __forceinline__ void online_update(v8f& sA, v8f& sB,
                                              float* m, float* l, v8f* o) {
#pragma unroll
    for (int r = 0; r < 8; ++r) {
        float sa = sA[r], sb = sB[r];
        float mx = fmaxf(sa, sb);
        mx = fmaxf(mx, __shfl_xor(mx, 1));
        mx = fmaxf(mx, __shfl_xor(mx, 2));
        mx = fmaxf(mx, __shfl_xor(mx, 4));
        mx = fmaxf(mx, __shfl_xor(mx, 8));
        float mnew  = fmaxf(m[r], mx);
        float alpha = __expf(m[r] - mnew);
        float pa = __expf(sa - mnew);
        float pb = __expf(sb - mnew);
        float rs = pa + pb;
        rs += __shfl_xor(rs, 1);
        rs += __shfl_xor(rs, 2);
        rs += __shfl_xor(rs, 4);
        rs += __shfl_xor(rs, 8);
        l[r] = l[r] * alpha + rs;
        m[r] = mnew;
        sA[r] = pa; sB[r] = pb;
#pragma unroll
        for (int t = 0; t < 4; ++t) o[t][r] *= alpha;
    }
}

// C layout (lane n = lane&15, rows r+8*(lane>=16)) -> row-major LDS tile.
__device__ __forceinline__ void store_p(__bf16 (*sp)[KT], const v8f& pa,
                                        const v8f& pb, int lane) {
    int nn  = lane & 15;
    int mof = (lane < 16) ? 0 : 8;
#pragma unroll
    for (int r = 0; r < 8; ++r) {
        sp[mof + r][nn]      = (__bf16)pa[r];
        sp[mof + r][nn + 16] = (__bf16)pb[r];
    }
}

// ---------------------------------------------------------------------------
// Projection: q = x Wq^T, k = x Wk^T + bk, v = x Wv^T, plus bf16 copy of x.
// ---------------------------------------------------------------------------
__global__ __launch_bounds__(256) void proj_kernel(
    const float* __restrict__ x,  const float* __restrict__ Wq,
    const float* __restrict__ Wk, const float* __restrict__ bk,
    const float* __restrict__ Wv,
    __bf16* __restrict__ xb, __bf16* __restrict__ qb,
    __bf16* __restrict__ kb, __bf16* __restrict__ vb) {
    int t = blockIdx.x * blockDim.x + threadIdx.x;
    if (t >= BB * SS * DD) return;
    int dout = t & (DD - 1);
    int bs   = t >> 6;                      // b*S + s
    const float* xr = x + (size_t)bs * DD;
    float aq = 0.f, ak = 0.f, av = 0.f;
#pragma unroll 8
    for (int e = 0; e < DD; ++e) {
        float xe = xr[e];
        aq = fmaf(xe, Wq[dout * DD + e], aq);
        ak = fmaf(xe, Wk[dout * DD + e], ak);
        av = fmaf(xe, Wv[dout * DD + e], av);
    }
    ak += bk[dout];
    xb[t] = (__bf16)xr[dout];
    qb[t] = (__bf16)aq;
    kb[t] = (__bf16)ak;
    vb[t] = (__bf16)av;
}

// ---------------------------------------------------------------------------
// Flash causal dot-product attention (used for branches 1 and 2).
// One block = 64 queries of one batch (4 waves x 16 queries).
// accum=0: out = w*attn;  accum=1: out += w*attn.
// ---------------------------------------------------------------------------
__global__ __launch_bounds__(128) void dot_attn_kernel(
    const __bf16* __restrict__ Qb, const __bf16* __restrict__ Kb,
    const __bf16* __restrict__ Vb,
    const float* __restrict__ attn_w, int widx,
    const float* __restrict__ attn_scale, int use_scale,
    float* __restrict__ out, int accum) {
    __shared__ __bf16 sK [KT][DD];         // keys, row-major   (QK^T B-frags)
    __shared__ __bf16 sVT[DD][KT];         // values, transposed (P·V B-frags)
    __shared__ __bf16 sP [NW][16][KT];     // P staging (C->A reshape)

    const int tid  = threadIdx.x;
    const int wave = tid >> 5;
    const int lane = tid & 31;
    const int b    = blockIdx.x / (SS / BQ);
    const int q0b  = (blockIdx.x % (SS / BQ)) * BQ;
    const int qbase = q0b + wave * 16;
    const int nn   = lane & 15;
    const int mof  = (lane < 16) ? 0 : 8;

    const __bf16* qqp = Qb + ((size_t)(b * SS + qbase)) * DD;
    v16bf aQ0 = wmma_a_frag(qqp,      DD, lane);
    v16bf aQ1 = wmma_a_frag(qqp + 32, DD, lane);

    v8f o[4] = {};
    float m[8], l[8];
#pragma unroll
    for (int r = 0; r < 8; ++r) { m[r] = -3.0e38f; l[r] = 0.f; }

    const float sc = use_scale ? (0.125f * attn_scale[0]) : 1.0f;
    const int nkg = (q0b + BQ) / KT;

    for (int kg = 0; kg < nkg; ++kg) {
        const int k0 = kg * KT;
        __syncthreads();
        for (int i = tid; i < KT * DD; i += 128) {
            int key = i >> 6, d = i & (DD - 1);
            size_t gi = ((size_t)(b * SS + k0 + key)) * DD + d;
            sK [key][d] = Kb[gi];
            sVT[d][key] = Vb[gi];
            if (k0 + KT < SS)
                __builtin_prefetch(&Kb[gi + (size_t)KT * DD], 0, 1);
        }
        __syncthreads();

        if (k0 > qbase + 15) continue;   // fully-masked for this wave

        // scores: 16x32 per wave = two C tiles, 4 WMMAs
        v8f sa = {}, sb = {};
        {
            v16bf b00 = wmma_b_frag_rm(&sK[0][0], DD,  0,  0, lane);
            v16bf b01 = wmma_b_frag_rm(&sK[0][0], DD,  0, 32, lane);
            v16bf b10 = wmma_b_frag_rm(&sK[0][0], DD, 16,  0, lane);
            v16bf b11 = wmma_b_frag_rm(&sK[0][0], DD, 16, 32, lane);
            sa = wmma_bf16(aQ0, b00, sa); sa = wmma_bf16(aQ1, b01, sa);
            sb = wmma_bf16(aQ0, b10, sb); sb = wmma_bf16(aQ1, b11, sb);
        }

        // scale + causal mask
#pragma unroll
        for (int r = 0; r < 8; ++r) {
            int q = qbase + mof + r;
            sa[r] *= sc; sb[r] *= sc;
            if (k0 + nn      > q) sa[r] = -1e9f;
            if (k0 + 16 + nn > q) sb[r] = -1e9f;
        }

        online_update(sa, sb, m, l, o);

        // P·V: reshape via per-wave LDS, then 4 WMMAs
        store_p(sP[wave], sa, sb, lane);
        lds_fence_wave();
        v16bf p = wmma_a_frag(&sP[wave][0][0], KT, lane);
#pragma unroll
        for (int t = 0; t < 4; ++t) {
            v16bf bv = wmma_b_frag_rm(&sVT[0][0], KT, t * 16, 0, lane);
            o[t] = wmma_bf16(p, bv, o[t]);
        }
    }

    const float wsum = attn_w[0] + attn_w[1] + attn_w[2];
    const float w    = attn_w[widx] / wsum;
#pragma unroll
    for (int t = 0; t < 4; ++t) {
#pragma unroll
        for (int r = 0; r < 8; ++r) {
            int q = qbase + mof + r;
            size_t oi = ((size_t)(b * SS + q)) * DD + t * 16 + nn;
            float val = w * o[t][r] / l[r];
            if (accum) val += out[oi];
            out[oi] = val;
        }
    }
}

// ---------------------------------------------------------------------------
// Flash causal additive (Bahdanau) attention: s[q,k] = sum_d tanh(x_q + x_k).
// Trans-op bound (B*S*S*D tanh) -> kept lean so occupancy stays high.
// Always accumulates: out += w2 * o3.
// ---------------------------------------------------------------------------
__global__ __launch_bounds__(128) void add_attn_kernel(
    const float* __restrict__ x, const __bf16* __restrict__ xb,
    const float* __restrict__ attn_w, float* __restrict__ out) {
    __shared__ float  sXkf[KT][DD];        // x keys, f32 (tanh scores)
    __shared__ __bf16 sXkT[DD][KT];        // x keys, transposed (P·x B-frags)
    __shared__ float  sXqf[NW][16][DD];    // x queries, f32
    __shared__ __bf16 sP  [NW][16][KT];    // P staging

    const int tid  = threadIdx.x;
    const int wave = tid >> 5;
    const int lane = tid & 31;
    const int b    = blockIdx.x / (SS / BQ);
    const int q0b  = (blockIdx.x % (SS / BQ)) * BQ;
    const int qbase = q0b + wave * 16;
    const int nn   = lane & 15;
    const int mof  = (lane < 16) ? 0 : 8;

    for (int i = lane; i < 16 * DD; i += 32) {
        int r = i >> 6, c = i & (DD - 1);
        sXqf[wave][r][c] = x[((size_t)(b * SS + qbase + r)) * DD + c];
    }

    v8f o[4] = {};
    float m[8], l[8];
#pragma unroll
    for (int r = 0; r < 8; ++r) { m[r] = -3.0e38f; l[r] = 0.f; }

    const int nkg = (q0b + BQ) / KT;

    for (int kg = 0; kg < nkg; ++kg) {
        const int k0 = kg * KT;
        __syncthreads();
        for (int i = tid; i < KT * DD; i += 128) {
            int key = i >> 6, d = i & (DD - 1);
            size_t gi = ((size_t)(b * SS + k0 + key)) * DD + d;
            sXkf[key][d] = x[gi];
            sXkT[d][key] = xb[gi];
            if (k0 + KT < SS)
                __builtin_prefetch(&x[gi + (size_t)KT * DD], 0, 1);
        }
        __syncthreads();

        if (k0 > qbase + 15) continue;

        // additive scores: tanh reduction over D (trans-op hot loop)
        v8f sa = {}, sb = {};
#pragma unroll 4
        for (int d = 0; d < DD; ++d) {
            float xk0 = sXkf[nn][d];
            float xk1 = sXkf[nn + 16][d];
#pragma unroll
            for (int r = 0; r < 8; ++r) {
                float xq = sXqf[wave][mof + r][d];
                sa[r] += dev_tanh(xq + xk0);
                sb[r] += dev_tanh(xq + xk1);
            }
        }

#pragma unroll
        for (int r = 0; r < 8; ++r) {
            int q = qbase + mof + r;
            if (k0 + nn      > q) sa[r] = -1e9f;
            if (k0 + 16 + nn > q) sb[r] = -1e9f;
        }

        online_update(sa, sb, m, l, o);

        store_p(sP[wave], sa, sb, lane);
        lds_fence_wave();
        v16bf p = wmma_a_frag(&sP[wave][0][0], KT, lane);
#pragma unroll
        for (int t = 0; t < 4; ++t) {
            v16bf bx = wmma_b_frag_rm(&sXkT[0][0], KT, t * 16, 0, lane);
            o[t] = wmma_bf16(p, bx, o[t]);
        }
    }

    const float wsum = attn_w[0] + attn_w[1] + attn_w[2];
    const float w2   = attn_w[2] / wsum;
#pragma unroll
    for (int t = 0; t < 4; ++t) {
#pragma unroll
        for (int r = 0; r < 8; ++r) {
            int q = qbase + mof + r;
            size_t oi = ((size_t)(b * SS + q)) * DD + t * 16 + nn;
            out[oi] += w2 * o[t][r] / l[r];
        }
    }
}

extern "C" void kernel_launch(void* const* d_in, const int* in_sizes, int n_in,
                              void* d_out, int out_size, void* d_ws, size_t ws_size,
                              hipStream_t stream) {
    const float* x          = (const float*)d_in[0];
    const float* Wq         = (const float*)d_in[1];
    const float* Wk         = (const float*)d_in[2];
    const float* bk         = (const float*)d_in[3];
    const float* Wv         = (const float*)d_in[4];
    const float* attn_w     = (const float*)d_in[5];
    const float* attn_scale = (const float*)d_in[6];
    float* out = (float*)d_out;

    const size_t N = (size_t)BB * SS * DD;
    __bf16* xb = (__bf16*)d_ws;
    __bf16* qb = xb + N;
    __bf16* kb = xb + 2 * N;
    __bf16* vb = xb + 3 * N;

    proj_kernel<<<(int)((N + 255) / 256), 256, 0, stream>>>(
        x, Wq, Wk, bk, Wv, xb, qb, kb, vb);

    const int grid = BB * (SS / BQ);
    // branch 1: plain causal self-attention on raw x (writes out)
    dot_attn_kernel<<<grid, 128, 0, stream>>>(
        xb, xb, xb, attn_w, 0, attn_scale, 0, out, 0);
    // branch 2: scaled dot-product attention on q/k/v (accumulates)
    dot_attn_kernel<<<grid, 128, 0, stream>>>(
        qb, kb, vb, attn_w, 1, attn_scale, 1, out, 1);
    // branch 3: additive tanh attention (accumulates)
    add_attn_kernel<<<grid, 128, 0, stream>>>(
        x, xb, attn_w, out);
}
